// MHSA_266287972557
// MI455X (gfx1250) — matrix operand order repaired and checked by reading, not compiled, not run on hardware
//
#include <hip/hip_runtime.h>
#include <hip/hip_bf16.h>

typedef __attribute__((ext_vector_type(16))) _Float16 v16h;
typedef __attribute__((ext_vector_type(8)))  _Float16 v8h;
typedef __attribute__((ext_vector_type(4)))  _Float16 v4h;
typedef __attribute__((ext_vector_type(8)))  float    v8f;
typedef __attribute__((ext_vector_type(4)))  float    v4f;

#define SEQ_N    4096
#define DIM_C    256
#define HEADS    4
#define HDIM     64
#define BATCH    2
#define MROWS    (BATCH * SEQ_N)   /* 8192 flattened (b,n) rows */
#define QKV_COLS (3 * DIM_C)       /* 768 */
#define SCALE_F  0.125f            /* 64^-0.5 */

#define XN  (MROWS * DIM_C)        /* 2,097,152 */
#define WQN (QKV_COLS * DIM_C)     /*   196,608 */
#define WPN (DIM_C * DIM_C)        /*    65,536 */

__device__ __forceinline__ v8f wmma_f16(v16h a, v16h b, v8f c) {
  // D = A(16x32 f16) x B(32x16 f16) + C(16x16 f32)
  return __builtin_amdgcn_wmma_f32_16x16x32_f16(false, a, false, b, (short)0, c,
                                                false, false);
}

// ---- operand builders (this file's contiguous-k layout: k = 16*half + j) --------------------------------
// A-matrix 16x32 f16: lane holds row (lane&15); elems j = k+2*half8+j,
// half8 = (lane>=16)?8:0.
__device__ __forceinline__ v16h mk_a_f16(const _Float16* row, int kb, int half8) {
  v8h lo = *(const v8h*)(row + kb + 2 * half8);
  v8h hi = *(const v8h*)(row + kb + 2 * half8 + 8);
  v16h r;
#pragma unroll
  for (int j = 0; j < 8; ++j) { r[j] = lo[j]; r[8 + j] = hi[j]; }
  return r;
}
// B-matrix 32x16 f16: lane holds col (lane&15); elems j = k+half16+j.
// Source rows have the contraction dim contiguous -> one 32B load.
__device__ __forceinline__ v16h ld_b_f16(const _Float16* rowc, int kb, int half16) {
  return *(const v16h*)(rowc + kb + half16);
}

// ===========================================================================
// Kernel 0: one-time f32 -> f16 conversion of x, w_qkv, w_proj.
// ===========================================================================
__global__ void cvt_kernel(const float* __restrict__ x,
                           const float* __restrict__ wq,
                           const float* __restrict__ wp,
                           _Float16* __restrict__ xh,
                           _Float16* __restrict__ wqh,
                           _Float16* __restrict__ wph) {
  const size_t i = (size_t)(blockIdx.x * blockDim.x + threadIdx.x) * 4;
  const float* src;
  _Float16* dst;
  size_t off;
  if (i < XN)            { src = x;  dst = xh;  off = i; }
  else if (i < XN + WQN) { src = wq; dst = wqh; off = i - XN; }
  else                   { src = wp; dst = wph; off = i - XN - WQN; }
  const v4f v = *(const v4f*)(src + off);
  v4h h;
#pragma unroll
  for (int j = 0; j < 4; ++j) h[j] = (_Float16)v[j];
  *(v4h*)(dst + off) = h;
}

// ===========================================================================
// Kernel 1: qkv = x @ w_qkv^T  (M=8192, N=768, K=256), f16 operands.
// One wave -> 32x64 tile: 2 A tiles x 4 B tiles -> 8 WMMAs per k-step from
// 6 tile loads.  256*12 = 3072 waves, 8 waves/block.
// Writes Q,K as [B,H,N,D] f16 and V transposed as [B,H,D,N] f16.
// ===========================================================================
__global__ void qkv_kernel(const _Float16* __restrict__ xh,
                           const _Float16* __restrict__ wqh,
                           _Float16* __restrict__ Qo,
                           _Float16* __restrict__ Ko,
                           _Float16* __restrict__ Vt) {
  const int lane  = threadIdx.x & 31;
  const int wave  = threadIdx.x >> 5;
  const int id    = blockIdx.x * (blockDim.x >> 5) + wave;
  const int mt    = id / (QKV_COLS / 64);
  const int ng    = id % (QKV_COLS / 64);
  const int m0    = mt * 32, c0 = ng * 64;
  const int col   = lane & 15;
  const int half8 = (lane >> 4) << 3;
  const int half16= (lane >> 4) << 4;

  const _Float16* ar0 = xh + (size_t)(m0 + col) * DIM_C;
  const _Float16* ar1 = xh + (size_t)(m0 + 16 + col) * DIM_C;
  const _Float16* br0 = wqh + (size_t)(c0 +  0 + col) * DIM_C;
  const _Float16* br1 = wqh + (size_t)(c0 + 16 + col) * DIM_C;
  const _Float16* br2 = wqh + (size_t)(c0 + 32 + col) * DIM_C;
  const _Float16* br3 = wqh + (size_t)(c0 + 48 + col) * DIM_C;

  v8f acc[2][4] = {};
#pragma unroll
  for (int kb = 0; kb < DIM_C; kb += 32) {
    const v16h a0 = mk_a_f16(ar0, kb, half8);
    const v16h a1 = mk_a_f16(ar1, kb, half8);
    const v16h b0 = ld_b_f16(br0, kb, half16);
    const v16h b1 = ld_b_f16(br1, kb, half16);
    const v16h b2 = ld_b_f16(br2, kb, half16);
    const v16h b3 = ld_b_f16(br3, kb, half16);
    acc[0][0] = wmma_f16(a0, b0, acc[0][0]);
    acc[0][1] = wmma_f16(a0, b1, acc[0][1]);
    acc[0][2] = wmma_f16(a0, b2, acc[0][2]);
    acc[0][3] = wmma_f16(a0, b3, acc[0][3]);
    acc[1][0] = wmma_f16(a1, b0, acc[1][0]);
    acc[1][1] = wmma_f16(a1, b1, acc[1][1]);
    acc[1][2] = wmma_f16(a1, b2, acc[1][2]);
    acc[1][3] = wmma_f16(a1, b3, acc[1][3]);
  }

#pragma unroll
  for (int s = 0; s < 4; ++s) {
    const int ccol  = c0 + s * 16 + col;
    const int which = ccol >> 8;        // 0=q 1=k 2=v (uniform per subtile)
    const int cc    = ccol & 255;
    const int h     = cc >> 6;          // uniform per subtile
    const int d     = cc & 63;
#pragma unroll
    for (int r = 0; r < 2; ++r) {
#pragma unroll
      for (int i = 0; i < 8; ++i) {
        const int mrow = m0 + r * 16 + half8 + i;
        const int b    = mrow >> 12;
        const int n    = mrow & (SEQ_N - 1);
        const _Float16 val = (_Float16)acc[r][s][i];
        if (which == 0)
          Qo[(((size_t)(b * HEADS + h)) * SEQ_N + n) * HDIM + d] = val;
        else if (which == 1)
          Ko[(((size_t)(b * HEADS + h)) * SEQ_N + n) * HDIM + d] = val;
        else
          Vt[(((size_t)(b * HEADS + h)) * HDIM + d) * SEQ_N + n] = val;
      }
    }
  }
}

// ===========================================================================
// Kernel 2: flash attention, one wave per 32 query rows of one (b,h).
// S^T = K_chunk @ Q^T (K A-tiles reused across both q-tiles);
// O^T = V^T @ P^T (V A-tiles reused across both q-tiles).
// 16 WMMAs per 32-key chunk from 8 tile loads.
// 2*4*128 = 1024 waves, 4 waves/block.
// ===========================================================================
__global__ void attn_kernel(const _Float16* __restrict__ Q,
                            const _Float16* __restrict__ Km,
                            const _Float16* __restrict__ Vt,
                            _Float16* __restrict__ AO) {
  const int lane  = threadIdx.x & 31;
  const int wave  = threadIdx.x >> 5;
  const int flat  = blockIdx.x * (blockDim.x >> 5) + wave;
  const int qt    = flat & (SEQ_N / 32 - 1);   // 0..127
  const int bh    = flat >> 7;                 // 0..7
  const int q0    = qt * 32;
  const int col   = lane & 15;
  const int half8 = (lane >> 4) << 3;
  const int half16= (lane >> 4) << 4;
  const bool hiH  = lane >= 16;

  const _Float16* q  = Q  + (size_t)bh * SEQ_N * HDIM;
  const _Float16* k  = Km + (size_t)bh * SEQ_N * HDIM;
  const _Float16* vt = Vt + (size_t)bh * HDIM * SEQ_N;

  // B operands for S^T: Q^T for both q-tiles, d-blocks [0,32) and [32,64).
  const _Float16* qra = q + (size_t)(q0 + col) * HDIM;
  const _Float16* qrb = q + (size_t)(q0 + 16 + col) * HDIM;
  const v16h bqa0 = *(const v16h*)(qra + half16);
  const v16h bqa1 = *(const v16h*)(qra + 32 + half16);
  const v16h bqb0 = *(const v16h*)(qrb + half16);
  const v16h bqb1 = *(const v16h*)(qrb + 32 + half16);

  v8f oa0 = {}, oa1 = {}, oa2 = {}, oa3 = {};
  v8f ob0 = {}, ob1 = {}, ob2 = {}, ob3 = {};
  float ma = -1e30f, la = 0.0f;
  float mb = -1e30f, lb = 0.0f;

  for (int kc = 0; kc < SEQ_N; kc += 32) {
    // V^T A-tiles for this chunk (independent of softmax): issue first
    const _Float16* vr0 = vt + (size_t)(0  + col) * SEQ_N + kc;
    const _Float16* vr1 = vt + (size_t)(16 + col) * SEQ_N + kc;
    const _Float16* vr2 = vt + (size_t)(32 + col) * SEQ_N + kc;
    const _Float16* vr3 = vt + (size_t)(48 + col) * SEQ_N + kc;
    const v16h av0 = mk_a_f16(vr0, 0, half8);
    const v16h av1 = mk_a_f16(vr1, 0, half8);
    const v16h av2 = mk_a_f16(vr2, 0, half8);
    const v16h av3 = mk_a_f16(vr3, 0, half8);

    // K A-tiles for this chunk
    const _Float16* kr0 = k + (size_t)(kc + col) * HDIM;
    const _Float16* kr1 = k + (size_t)(kc + 16 + col) * HDIM;
    const v16h ak0 = mk_a_f16(kr0, 0, half8);
    const v16h ak1 = mk_a_f16(kr0, 32, half8);
    const v16h ak2 = mk_a_f16(kr1, 0, half8);
    const v16h ak3 = mk_a_f16(kr1, 32, half8);

    // prefetch next chunk (gfx1250 global_prefetch)
    __builtin_prefetch(kr0 + 32 * HDIM, 0, 3);
    __builtin_prefetch(vr0 + 32, 0, 3);
    __builtin_prefetch(vr2 + 32, 0, 3);

    // ---- S^T = K_chunk @ Q^T for both q-tiles ----
    v8f sa0 = {}, sa1 = {}, sb0 = {}, sb1 = {};
    sa0 = wmma_f16(ak0, bqa0, sa0);
    sa0 = wmma_f16(ak1, bqa1, sa0);
    sa1 = wmma_f16(ak2, bqa0, sa1);
    sa1 = wmma_f16(ak3, bqa1, sa1);
    sb0 = wmma_f16(ak0, bqb0, sb0);
    sb0 = wmma_f16(ak1, bqb1, sb0);
    sb1 = wmma_f16(ak2, bqb0, sb1);
    sb1 = wmma_f16(ak3, bqb1, sb1);

    union H8U { v8h h; int u[4]; };

    // ---- online softmax + pack, q-tile a ----
    {
      float mx = -1e30f;
#pragma unroll
      for (int i = 0; i < 8; ++i) {
        sa0[i] *= SCALE_F; sa1[i] *= SCALE_F;
        mx = fmaxf(mx, fmaxf(sa0[i], sa1[i]));
      }
      mx = fmaxf(mx, __shfl_xor(mx, 16, 32));
      const float mnew  = fmaxf(ma, mx);
      const float alpha = __expf(ma - mnew);
      float rs = 0.0f;
#pragma unroll
      for (int i = 0; i < 8; ++i) {
        sa0[i] = __expf(sa0[i] - mnew); rs += sa0[i];
        sa1[i] = __expf(sa1[i] - mnew); rs += sa1[i];
      }
      rs += __shfl_xor(rs, 16, 32);
      la = la * alpha + rs;
      ma = mnew;
#pragma unroll
      for (int i = 0; i < 8; ++i) { oa0[i]*=alpha; oa1[i]*=alpha; oa2[i]*=alpha; oa3[i]*=alpha; }

      H8U p0, p1, s0, s1;
#pragma unroll
      for (int i = 0; i < 8; ++i) { p0.h[i] = (_Float16)sa0[i]; p1.h[i] = (_Float16)sa1[i]; }
#pragma unroll
      for (int j = 0; j < 4; ++j) {
        s0.u[j] = __shfl_xor(p0.u[j], 16, 32);
        s1.u[j] = __shfl_xor(p1.u[j], 16, 32);
      }
      const v8h lowv = hiH ? s1.h : p0.h;
      const v8h higv = hiH ? p1.h : s0.h;
      v16h bpa;
#pragma unroll
      for (int j = 0; j < 8; ++j) { bpa[j] = lowv[j]; bpa[8 + j] = higv[j]; }

      oa0 = wmma_f16(av0, bpa, oa0);
      oa1 = wmma_f16(av1, bpa, oa1);
      oa2 = wmma_f16(av2, bpa, oa2);
      oa3 = wmma_f16(av3, bpa, oa3);
    }

    // ---- online softmax + pack, q-tile b ----
    {
      float mx = -1e30f;
#pragma unroll
      for (int i = 0; i < 8; ++i) {
        sb0[i] *= SCALE_F; sb1[i] *= SCALE_F;
        mx = fmaxf(mx, fmaxf(sb0[i], sb1[i]));
      }
      mx = fmaxf(mx, __shfl_xor(mx, 16, 32));
      const float mnew  = fmaxf(mb, mx);
      const float alpha = __expf(mb - mnew);
      float rs = 0.0f;
#pragma unroll
      for (int i = 0; i < 8; ++i) {
        sb0[i] = __expf(sb0[i] - mnew); rs += sb0[i];
        sb1[i] = __expf(sb1[i] - mnew); rs += sb1[i];
      }
      rs += __shfl_xor(rs, 16, 32);
      lb = lb * alpha + rs;
      mb = mnew;
#pragma unroll
      for (int i = 0; i < 8; ++i) { ob0[i]*=alpha; ob1[i]*=alpha; ob2[i]*=alpha; ob3[i]*=alpha; }

      H8U p0, p1, s0, s1;
#pragma unroll
      for (int i = 0; i < 8; ++i) { p0.h[i] = (_Float16)sb0[i]; p1.h[i] = (_Float16)sb1[i]; }
#pragma unroll
      for (int j = 0; j < 4; ++j) {
        s0.u[j] = __shfl_xor(p0.u[j], 16, 32);
        s1.u[j] = __shfl_xor(p1.u[j], 16, 32);
      }
      const v8h lowv = hiH ? s1.h : p0.h;
      const v8h higv = hiH ? p1.h : s0.h;
      v16h bpb;
#pragma unroll
      for (int j = 0; j < 8; ++j) { bpb[j] = lowv[j]; bpb[8 + j] = higv[j]; }

      ob0 = wmma_f16(av0, bpb, ob0);
      ob1 = wmma_f16(av1, bpb, ob1);
      ob2 = wmma_f16(av2, bpb, ob2);
      ob3 = wmma_f16(av3, bpb, ob3);
    }
  }

  // ---- epilogue: divide by lsum, store f16 attn-out as [B,N,C] ----
  const int b = bh >> 2, h = bh & 3;
  {
    const float inv = 1.0f / la;
    _Float16* orow = AO + ((size_t)b * SEQ_N + (q0 + col)) * DIM_C + h * HDIM;
#pragma unroll
    for (int i = 0; i < 8; ++i) {
      orow[half8 + i]      = (_Float16)(oa0[i] * inv);
      orow[16 + half8 + i] = (_Float16)(oa1[i] * inv);
      orow[32 + half8 + i] = (_Float16)(oa2[i] * inv);
      orow[48 + half8 + i] = (_Float16)(oa3[i] * inv);
    }
  }
  {
    const float inv = 1.0f / lb;
    _Float16* orow = AO + ((size_t)b * SEQ_N + (q0 + 16 + col)) * DIM_C + h * HDIM;
#pragma unroll
    for (int i = 0; i < 8; ++i) {
      orow[half8 + i]      = (_Float16)(ob0[i] * inv);
      orow[16 + half8 + i] = (_Float16)(ob1[i] * inv);
      orow[32 + half8 + i] = (_Float16)(ob2[i] * inv);
      orow[48 + half8 + i] = (_Float16)(ob3[i] * inv);
    }
  }
}

// ===========================================================================
// Kernel 3: out = attn @ w_proj^T + b_proj (M=8192, N=256, K=256), f32 out.
// One wave -> 32x64 tile.  256*4 = 1024 waves, 8 waves/block.
// ===========================================================================
__global__ void proj_kernel(const _Float16* __restrict__ A,
                            const _Float16* __restrict__ wph,
                            const float* __restrict__ b_proj,
                            float* __restrict__ out) {
  const int lane  = threadIdx.x & 31;
  const int wave  = threadIdx.x >> 5;
  const int id    = blockIdx.x * (blockDim.x >> 5) + wave;
  const int mt    = id / (DIM_C / 64);
  const int ng    = id % (DIM_C / 64);
  const int m0    = mt * 32, c0 = ng * 64;
  const int col   = lane & 15;
  const int half8 = (lane >> 4) << 3;
  const int half16= (lane >> 4) << 4;

  const _Float16* ar0 = A + (size_t)(m0 + col) * DIM_C;
  const _Float16* ar1 = A + (size_t)(m0 + 16 + col) * DIM_C;
  const _Float16* br0 = wph + (size_t)(c0 +  0 + col) * DIM_C;
  const _Float16* br1 = wph + (size_t)(c0 + 16 + col) * DIM_C;
  const _Float16* br2 = wph + (size_t)(c0 + 32 + col) * DIM_C;
  const _Float16* br3 = wph + (size_t)(c0 + 48 + col) * DIM_C;

  v8f acc[2][4] = {};
#pragma unroll
  for (int kb = 0; kb < DIM_C; kb += 32) {
    const v16h a0 = mk_a_f16(ar0, kb, half8);
    const v16h a1 = mk_a_f16(ar1, kb, half8);
    const v16h b0 = ld_b_f16(br0, kb, half16);
    const v16h b1 = ld_b_f16(br1, kb, half16);
    const v16h b2 = ld_b_f16(br2, kb, half16);
    const v16h b3 = ld_b_f16(br3, kb, half16);
    acc[0][0] = wmma_f16(a0, b0, acc[0][0]);
    acc[0][1] = wmma_f16(a0, b1, acc[0][1]);
    acc[0][2] = wmma_f16(a0, b2, acc[0][2]);
    acc[0][3] = wmma_f16(a0, b3, acc[0][3]);
    acc[1][0] = wmma_f16(a1, b0, acc[1][0]);
    acc[1][1] = wmma_f16(a1, b1, acc[1][1]);
    acc[1][2] = wmma_f16(a1, b2, acc[1][2]);
    acc[1][3] = wmma_f16(a1, b3, acc[1][3]);
  }

#pragma unroll
  for (int s = 0; s < 4; ++s) {
    const int   ccol = c0 + s * 16 + col;
    const float bias = b_proj[ccol];
#pragma unroll
    for (int r = 0; r < 2; ++r) {
#pragma unroll
      for (int i = 0; i < 8; ++i)
        out[(size_t)(m0 + r * 16 + half8 + i) * DIM_C + ccol] = acc[r][s][i] + bias;
    }
  }
}

// ===========================================================================
extern "C" void kernel_launch(void* const* d_in, const int* in_sizes, int n_in,
                              void* d_out, int out_size, void* d_ws, size_t ws_size,
                              hipStream_t stream) {
  (void)in_sizes; (void)n_in; (void)out_size; (void)ws_size;
  const float* x      = (const float*)d_in[0];
  const float* w_qkv  = (const float*)d_in[1];
  const float* w_proj = (const float*)d_in[2];
  const float* b_proj = (const float*)d_in[3];
  float*       out    = (float*)d_out;

  char* ws = (char*)d_ws;
  // xh (4 MB) is dead after qkv_kernel; attention output AO aliases it.
  _Float16* xh  = (_Float16*)(ws);                                  // [M,C] f16
  _Float16* AOw = (_Float16*)(ws);                                  // [B,N,C] f16
  _Float16* Qw  = (_Float16*)(ws + ((size_t)4  << 20));             // [B,H,N,D]
  _Float16* Kw  = (_Float16*)(ws + ((size_t)8  << 20));             // [B,H,N,D]
  _Float16* Vtw = (_Float16*)(ws + ((size_t)12 << 20));             // [B,H,D,N]
  _Float16* wqh = (_Float16*)(ws + ((size_t)16 << 20));             // [768,256]
  _Float16* wph = (_Float16*)(ws + ((size_t)16 << 20) + (512 << 10)); // [256,256]

  cvt_kernel<<<(XN + WQN + WPN) / 4 / 256, 256, 0, stream>>>(
      x, w_qkv, w_proj, xh, wqh, wph);
  // 256*12 tiles of 32x64, 8 waves/block
  qkv_kernel<<<(MROWS / 32) * (QKV_COLS / 64) / 8, 256, 0, stream>>>(
      xh, wqh, Qw, Kw, Vtw);
  // 1024 waves, 4 waves/block
  attn_kernel<<<(BATCH * HEADS * (SEQ_N / 32)) / 4, 128, 0, stream>>>(
      Qw, Kw, Vtw, AOw);
  // 256*4 tiles of 32x64, 8 waves/block
  proj_kernel<<<(MROWS / 32) * (DIM_C / 64) / 8, 256, 0, stream>>>(
      AOw, wph, b_proj, out);
}
